// SharedSelfAttentionProcessor_22814866276359
// MI455X (gfx1250) — compile-verified, matrix operand order checked by reading, not run on hardware
//
#include <hip/hip_runtime.h>
#include <hip/hip_bf16.h>

#define BB 2
#define SS 2048
#define DD 640
#define HH 10
#define HD 64
#define S2 (2*SS)
#define EPSF 1e-5f
#define NEGF -1e9f

typedef __attribute__((ext_vector_type(16))) _Float16 v16h;
typedef __attribute__((ext_vector_type(8)))  float    v8f;
typedef __attribute__((ext_vector_type(4)))  int      v4i;

#if defined(__has_builtin)
#if __has_builtin(__builtin_amdgcn_global_load_async_to_lds_b128) && \
    __has_builtin(__builtin_amdgcn_s_wait_asynccnt)
#define USE_ASYNC_LDS 1
#endif
#endif

// Copy 16 bytes global -> LDS. Prefer CDNA5 async DMA path (ASYNCcnt) if the
// toolchain declares it; otherwise a b128 load + ds_store pair.
__device__ __forceinline__ void copy16_g2l(const _Float16* gsrc, _Float16* ldst) {
#if USE_ASYNC_LDS
  __builtin_amdgcn_global_load_async_to_lds_b128(
      (__attribute__((address_space(1))) v4i*)gsrc,
      (__attribute__((address_space(3))) v4i*)ldst, 0, 0);
#else
  *(uint4*)ldst = *(const uint4*)gsrc;
#endif
}

__device__ __forceinline__ void wait_g2l() {
#if USE_ASYNC_LDS
  __builtin_amdgcn_s_wait_asynccnt(0);
#endif
}

// Load a 16x32 f16 fragment (A row-major rows, or B as column-major columns).
// Per CDNA5 ISA 7.12.2: lane<16 -> idx=lane, K{0..7,16..23}; lane>=16 -> idx=lane-16, K{8..15,24..31}.
__device__ __forceinline__ v16h ld_frag(const _Float16* base, int ld) {
  const int lane = threadIdx.x & 31;
  const int r  = lane & 15;
  const int ko = (lane >> 4) << 3;          // 0 or 8
  const unsigned* p0 = (const unsigned*)(base + r * ld + ko);
  const unsigned* p1 = (const unsigned*)(base + r * ld + ko + 16);
  union { v16h v; unsigned u[8]; } f;
  f.u[0] = p0[0]; f.u[1] = p0[1]; f.u[2] = p0[2]; f.u[3] = p0[3];
  f.u[4] = p1[0]; f.u[5] = p1[1]; f.u[6] = p1[2]; f.u[7] = p1[3];
  return f.v;
}

__device__ __forceinline__ v8f wmma16(v16h a, v16h b, v8f c) {
  return __builtin_amdgcn_wmma_f32_16x16x32_f16(false, a, false, b, (short)0, c, false, false);
}

// Stage a float tile (rows x 32) as f16 into LDS: float4 global loads, uint2 LDS stores.
__device__ __forceinline__ void stage_f32_tile(const float* __restrict__ src, int ldsrc,
                                               int k0, _Float16* dst, int n4, int tid) {
  for (int i = 0; i < n4; ++i) {
    int idx = tid + i * 256;
    int r = idx >> 3, c4 = idx & 7;
    float4 v = *(const float4*)(src + r * ldsrc + k0 + c4 * 4);
    union { _Float16 h[4]; uint2 u; } pk;
    pk.h[0] = (_Float16)v.x; pk.h[1] = (_Float16)v.y;
    pk.h[2] = (_Float16)v.z; pk.h[3] = (_Float16)v.w;
    *(uint2*)(dst + r * 32 + c4 * 4) = pk.u;
  }
}

// ---------------- Kernel 1: QKV projections  Y = X @ W^T ----------------
// grid: (M/128, D/64, 3)  block: 256 (8 waves, each a 16x64 strip)
__global__ __launch_bounds__(256) void proj_qkv_kernel(
    const float* __restrict__ X, const float* __restrict__ Wq,
    const float* __restrict__ Wk, const float* __restrict__ Wv,
    float* __restrict__ qkvF) {
  __shared__ _Float16 xs[128 * 32];
  __shared__ _Float16 wsm[64 * 32];
  const int t = blockIdx.z;
  const float* Wmat = (t == 0) ? Wq : (t == 1) ? Wk : Wv;
  float* Y = qkvF + (size_t)t * (BB * HH * SS * HD);
  const int m0 = blockIdx.x * 128;
  const int n0 = blockIdx.y * 64;
  const int tid = threadIdx.x;
  const int w = tid >> 5, lane = tid & 31;
  const int half = lane >> 4, nlo = lane & 15;
  const float* Xt = X + (size_t)m0 * DD;      // 32-bit offsets below
  const float* Wt = Wmat + (size_t)n0 * DD;
  v8f acc[4] = {};
  for (int kk = 0; kk < DD / 32; ++kk) {
    const int k0 = kk * 32;
    stage_f32_tile(Xt, DD, k0, xs, 4, tid);   // 128x32
    stage_f32_tile(Wt, DD, k0, wsm, 2, tid);  // 64x32
    __syncthreads();
    v16h a = ld_frag(xs + w * 16 * 32, 32);
#pragma unroll
    for (int f = 0; f < 4; ++f)
      acc[f] = wmma16(a, ld_frag(wsm + f * 16 * 32, 32), acc[f]);
    __syncthreads();
  }
#pragma unroll
  for (int f = 0; f < 4; ++f)
#pragma unroll
    for (int i = 0; i < 8; ++i) {
      int m = m0 + w * 16 + i + 8 * half;
      int n = n0 + f * 16 + nlo;
      int b = m >> 11, s = m & (SS - 1);
      int h = n >> 6, hd = n & 63;
      Y[(((b * HH + h) * SS + s) << 6) + hd] = acc[f][i];
    }
}

// ---------------- Kernel 2: AdaIN stats (mean + unbiased std over S) ----------------
__global__ __launch_bounds__(256) void adain_stats_kernel(
    const float* __restrict__ qkvF, float* __restrict__ meanB, float* __restrict__ stdB) {
  __shared__ float ssum[256], ssq[256];
  const int blk = blockIdx.x;
  const int t = blk / (BB * HH);
  const int rem = blk % (BB * HH);
  const int b = rem / HH, h = rem % HH;
  const float* base = qkvF + (size_t)(((t * BB + b) * HH + h) * SS) * HD;
  const int hd = threadIdx.x & 63, chunk = threadIdx.x >> 6;
  float sum = 0.f, sq = 0.f;
  for (int s = chunk * 512; s < chunk * 512 + 512; ++s) {
    float x = base[(s << 6) + hd];
    sum += x; sq += x * x;
  }
  ssum[threadIdx.x] = sum; ssq[threadIdx.x] = sq;
  __syncthreads();
  if (chunk == 0) {
    sum = ssum[hd] + ssum[hd + 64] + ssum[hd + 128] + ssum[hd + 192];
    sq  = ssq[hd]  + ssq[hd + 64]  + ssq[hd + 128]  + ssq[hd + 192];
    float mean = sum / (float)SS;
    float var  = (sq - (float)SS * mean * mean) / (float)(SS - 1);  // ddof=1
    int c = ((t * BB + b) * HH + h) * HD + hd;
    meanB[c] = mean;
    stdB[c]  = sqrtf(var + EPSF);
  }
}

// ---------------- Kernel 3: AdaIN apply + pack f16 ----------------
__global__ __launch_bounds__(256) void adain_apply_kernel(
    const float* __restrict__ qkvF, const float* __restrict__ meanB,
    const float* __restrict__ stdB, _Float16* __restrict__ qkv16) {
  const unsigned N = 3u * BB * HH * SS * HD;
  for (unsigned idx = blockIdx.x * blockDim.x + threadIdx.x; idx < N;
       idx += gridDim.x * blockDim.x) {
    int hd = (int)(idx & 63);
    unsigned tmp2 = idx >> 17;          // (t*B+b)*H + h
    int h = (int)(tmp2 % HH);
    unsigned tmp3 = tmp2 / HH;          // t*B + b
    int b = (int)(tmp3 % BB);
    int t = (int)(tmp3 / BB);
    int c  = ((t * BB + b) * HH + h) * HD + hd;
    int c0 = ((t * BB + 0) * HH + h) * HD + hd;  // style = batch 0
    float x = qkvF[idx];
    float y = (x - meanB[c]) / stdB[c] * stdB[c0] + meanB[c0];
    qkv16[idx] = (_Float16)y;
  }
}

// ---------------- Kernel 4: flash attention over concat(k, k[0]) ----------------
// grid: (S/64, H, B), block 128 (4 waves x 16 q-rows)
__global__ __launch_bounds__(128) void attn_kernel(
    const _Float16* __restrict__ qkv16, const unsigned char* __restrict__ mask,
    float* __restrict__ AO) {
  __shared__ _Float16 qs[64 * 64];
  __shared__ _Float16 ks[64 * 64];
  __shared__ _Float16 vts[64 * 72];   // transposed V, padded stride
  __shared__ _Float16 ps[64 * 64];
  const int b = blockIdx.z, h = blockIdx.y;
  const int s0 = blockIdx.x * 64;
  const int tid = threadIdx.x, w = tid >> 5, lane = tid & 31;
  const int half = lane >> 4, nlo = lane & 15;
  const unsigned seg = BB * HH * SS * HD;
  const _Float16* q16 = qkv16;
  const _Float16* k16 = qkv16 + seg;
  const _Float16* v16 = qkv16 + 2 * seg;
  const unsigned headBase = (unsigned)((b * HH + h) * SS) << 6;   // element offset
  const unsigned head0Base = (unsigned)(h * SS) << 6;             // batch-0 head

  {
    const _Float16* qp = q16 + headBase + (s0 << 6);
#pragma unroll
    for (int i = 0; i < 4; ++i) {
      int e = (tid + i * 128) * 8;                 // 512 x 16B
      copy16_g2l(qp + e, qs + e);
    }
  }
  wait_g2l();
  __syncthreads();
  v16h aq0 = ld_frag(qs + w * 16 * HD, HD);
  v16h aq1 = ld_frag(qs + w * 16 * HD + 32, HD);

  float mrow[8], lrow[8];
  v8f o[4] = {};
#pragma unroll
  for (int i = 0; i < 8; ++i) { mrow[i] = -3.0e38f; lrow[i] = 0.f; }
  const float scale = 0.125f;  // 1/sqrt(64)
  const unsigned maskRow0 = (unsigned)(s0 + w * 16 + 8 * half) * S2 + nlo;

  for (int kt = 0; kt < S2 / 64; ++kt) {
    const int jbase = kt * 64;
    const _Float16* kp;
    const _Float16* vp;
    if (jbase < SS) {  // first half: own batch
      kp = k16 + headBase + ((unsigned)jbase << 6);
      vp = v16 + headBase + ((unsigned)jbase << 6);
    } else {           // second half: concat_first -> batch 0
      kp = k16 + head0Base + ((unsigned)(jbase - SS) << 6);
      vp = v16 + head0Base + ((unsigned)(jbase - SS) << 6);
    }
    __builtin_prefetch(kp + 64 * HD, 0, 0);  // global_prefetch next K tile
    __syncthreads();                          // previous-iter LDS reads done
#pragma unroll
    for (int i = 0; i < 4; ++i) {            // K tile: 512 x 16B async
      int e = (tid + i * 128) * 8;
      copy16_g2l(kp + e, ks + e);
    }
#pragma unroll
    for (int i = 0; i < 4; ++i) {            // V^T tile: b128 load, strided ds stores
      int i4 = tid + i * 128;                // 512 chunks of 8 hd-values
      int key = i4 >> 3, hdc = (i4 & 7) * 8;
      union { uint4 u; _Float16 hh[8]; } vv;
      vv.u = *(const uint4*)(vp + (key << 6) + hdc);
#pragma unroll
      for (int j = 0; j < 8; ++j) vts[(hdc + j) * 72 + key] = vv.hh[j];
    }
    wait_g2l();
    __syncthreads();

    v8f sf[4];
#pragma unroll
    for (int f = 0; f < 4; ++f) {
      v8f c = {};
      c = wmma16(aq0, ld_frag(ks + f * 16 * HD, HD), c);
      c = wmma16(aq1, ld_frag(ks + f * 16 * HD + 32, HD), c);
      sf[f] = c;
    }
#pragma unroll
    for (int f = 0; f < 4; ++f)
#pragma unroll
      for (int i = 0; i < 8; ++i) {
        float v = sf[f][i] * scale;
        if (!mask[maskRow0 + (unsigned)i * S2 + jbase + f * 16]) v = NEGF;
        sf[f][i] = v;
      }
    float newm[8], corr[8];
#pragma unroll
    for (int i = 0; i < 8; ++i) {
      float rm = fmaxf(fmaxf(sf[0][i], sf[1][i]), fmaxf(sf[2][i], sf[3][i]));
      rm = fmaxf(rm, __shfl_xor(rm, 1, 32));
      rm = fmaxf(rm, __shfl_xor(rm, 2, 32));
      rm = fmaxf(rm, __shfl_xor(rm, 4, 32));
      rm = fmaxf(rm, __shfl_xor(rm, 8, 32));
      newm[i] = fmaxf(mrow[i], rm);
      corr[i] = __expf(mrow[i] - newm[i]);
    }
#pragma unroll
    for (int f = 0; f < 4; ++f)
#pragma unroll
      for (int i = 0; i < 8; ++i)
        sf[f][i] = __expf(sf[f][i] - newm[i]);
#pragma unroll
    for (int i = 0; i < 8; ++i) {
      float r = sf[0][i] + sf[1][i] + sf[2][i] + sf[3][i];
      r += __shfl_xor(r, 1, 32);
      r += __shfl_xor(r, 2, 32);
      r += __shfl_xor(r, 4, 32);
      r += __shfl_xor(r, 8, 32);
      lrow[i] = lrow[i] * corr[i] + r;
      mrow[i] = newm[i];
    }
#pragma unroll
    for (int f = 0; f < 4; ++f)
#pragma unroll
      for (int i = 0; i < 8; ++i)
        o[f][i] *= corr[i];
#pragma unroll
    for (int f = 0; f < 4; ++f)
#pragma unroll
      for (int i = 0; i < 8; ++i)
        ps[(w * 16 + i + 8 * half) * 64 + f * 16 + nlo] = (_Float16)sf[f][i];
    __syncthreads();
#pragma unroll
    for (int ks2 = 0; ks2 < 2; ++ks2) {
      v16h pa = ld_frag(ps + w * 16 * 64 + ks2 * 32, 64);
#pragma unroll
      for (int f = 0; f < 4; ++f)
        o[f] = wmma16(pa, ld_frag(vts + f * 16 * 72 + ks2 * 32, 72), o[f]);
    }
  }
  float inv[8];
#pragma unroll
  for (int i = 0; i < 8; ++i) inv[i] = 1.0f / lrow[i];
#pragma unroll
  for (int f = 0; f < 4; ++f)
#pragma unroll
    for (int i = 0; i < 8; ++i) {
      int row = s0 + w * 16 + i + 8 * half;
      int col = h * HD + f * 16 + nlo;
      AO[(b * SS + row) * DD + col] = o[f][i] * inv[i];
    }
}

// ---------------- Kernel 5: output projection + bias + residual ----------------
__global__ __launch_bounds__(256) void out_proj_kernel(
    const float* __restrict__ AO, const float* __restrict__ Wo,
    const float* __restrict__ bo, const float* __restrict__ resid,
    float* __restrict__ out) {
  __shared__ _Float16 xs[128 * 32];
  __shared__ _Float16 wsm[64 * 32];
  const int m0 = blockIdx.x * 128;
  const int n0 = blockIdx.y * 64;
  const int tid = threadIdx.x;
  const int w = tid >> 5, lane = tid & 31;
  const int half = lane >> 4, nlo = lane & 15;
  const float* At = AO + (size_t)m0 * DD;
  const float* Wt = Wo + (size_t)n0 * DD;
  v8f acc[4] = {};
  for (int kk = 0; kk < DD / 32; ++kk) {
    const int k0 = kk * 32;
    stage_f32_tile(At, DD, k0, xs, 4, tid);
    stage_f32_tile(Wt, DD, k0, wsm, 2, tid);
    __syncthreads();
    v16h a = ld_frag(xs + w * 16 * 32, 32);
#pragma unroll
    for (int f = 0; f < 4; ++f)
      acc[f] = wmma16(a, ld_frag(wsm + f * 16 * 32, 32), acc[f]);
    __syncthreads();
  }
#pragma unroll
  for (int f = 0; f < 4; ++f)
#pragma unroll
    for (int i = 0; i < 8; ++i) {
      int m = m0 + w * 16 + i + 8 * half;
      int n = n0 + f * 16 + nlo;
      out[m * DD + n] = acc[f][i] + bo[n] + resid[m * DD + n];
    }
}

extern "C" void kernel_launch(void* const* d_in, const int* in_sizes, int n_in,
                              void* d_out, int out_size, void* d_ws, size_t ws_size,
                              hipStream_t stream) {
  const float* hidden = (const float*)d_in[0];
  const float* Wq = (const float*)d_in[1];
  const float* Wk = (const float*)d_in[2];
  const float* Wv = (const float*)d_in[3];
  const float* Wo = (const float*)d_in[4];
  const float* bo = (const float*)d_in[5];
  const unsigned char* mask = (const unsigned char*)d_in[6];
  float* out = (float*)d_out;

  char* ws = (char*)d_ws;
  float*     qkvF  = (float*)(ws);                 // 31,457,280 B  [3][B][H][S][64] f32
  float*     meanB = (float*)(ws + 31457280);      //     15,360 B
  float*     stdB  = (float*)(ws + 31472640);      //     15,360 B
  _Float16*  qkv16 = (_Float16*)(ws + 31488000);   // 15,728,640 B  f16 packed
  float*     AO    = (float*)(ws + 47216640);      // 10,485,760 B  [B,S,D] f32
  (void)ws_size; (void)in_sizes; (void)n_in; (void)out_size;

  proj_qkv_kernel<<<dim3(BB * SS / 128, DD / 64, 3), 256, 0, stream>>>(hidden, Wq, Wk, Wv, qkvF);
  adain_stats_kernel<<<dim3(3 * BB * HH), 256, 0, stream>>>(qkvF, meanB, stdB);
  adain_apply_kernel<<<dim3(3840), 256, 0, stream>>>(qkvF, meanB, stdB, qkv16);
  attn_kernel<<<dim3(SS / 64, HH, BB), 128, 0, stream>>>(qkv16, mask, AO);
  out_proj_kernel<<<dim3(BB * SS / 128, DD / 64, 1), 256, 0, stream>>>(AO, Wo, bo, hidden, out);
}
